// AttModel_40235253629210
// MI455X (gfx1250) — matrix operand order repaired and checked by reading, not compiled, not run on hardware
//
#include <hip/hip_runtime.h>
#include <hip/hip_bf16.h>
#include <cstddef>

// ---------------------------------------------------------------------------
// Model constants (match reference)
// ---------------------------------------------------------------------------
#define HD     1024
#define NHEAD  16
#define DHEAD  64
#define NBLK   4
#define BB     64
#define LVV    196
#define LSS    64
#define LQQ    20
#define LTT    260          // LV + LS
#define NCLSS  3129
#define EMBB   300
#define FFH    4096         // 4*H

// ---------------------------------------------------------------------------
// WMMA types (CDNA5 / gfx1250, wave32)
// ---------------------------------------------------------------------------
typedef __attribute__((ext_vector_type(16))) __bf16 bf16x16;
typedef __attribute__((ext_vector_type(8)))  float  floatx8;
// dword-aligned 128-bit vector (W rows with N=3129 are only 4B aligned)
typedef float float4a4 __attribute__((ext_vector_type(4), aligned(4)));

union BfPack { bf16x16 v; unsigned int u[8]; };

// two f32 -> packed bf16x2 in one dword: round-to-nearest (ties away) via
// +0x8000, then a single v_perm_b32 grabs both high halves.
__device__ __forceinline__ unsigned int pack_bf16x2(float lo, float hi) {
  const unsigned int ul = __float_as_uint(lo) + 0x8000u;
  const unsigned int uh = __float_as_uint(hi) + 0x8000u;
  // byte sel: {uh[3],uh[2],ul[3],ul[2]}  (src0=uh -> sel 4..7, src1=ul -> sel 0..3)
  return __builtin_amdgcn_perm(uh, ul, 0x07060302u);
}

// ---------------------------------------------------------------------------
// Fragment build + 4x WMMA for one staged K-tile (CDNA5 ISA 7.12.2 layouts)
// ---------------------------------------------------------------------------
__device__ __forceinline__ void wmma_step(const unsigned short As[64][36],
                                          const unsigned short Bs[128][36],
                                          int wm, int wn, int lr, bool hi,
                                          floatx8 acc[2][2])
{
  bf16x16 afrag[2], bfrag[2];
#pragma unroll
  for (int mb = 0; mb < 2; ++mb) {
    const int m = wm + mb * 16 + lr;
    const int khalf = hi ? 8 : 0;
    BfPack u;
#pragma unroll
    for (int j = 0; j < 8; ++j) {
      const int kb = ((j < 4) ? (2 * j) : (2 * j + 8)) + khalf;
      u.u[j] = *(const unsigned int*)&As[m][kb];
    }
    afrag[mb] = u.v;
  }
#pragma unroll
  for (int nb = 0; nb < 2; ++nb) {
    const int n = wn + nb * 16 + lr;
    const int koff = hi ? 8 : 0;
    BfPack u;
#pragma unroll
    for (int j = 0; j < 8; ++j)
      u.u[j] = *(const unsigned int*)&Bs[n][((j < 4) ? (2 * j) : (2 * j + 8)) + koff];
    bfrag[nb] = u.v;
  }
#pragma unroll
  for (int mb = 0; mb < 2; ++mb)
#pragma unroll
    for (int nb = 0; nb < 2; ++nb)
      acc[mb][nb] = __builtin_amdgcn_wmma_f32_16x16x32_bf16(
          false, afrag[mb], false, bfrag[nb], (short)0, acc[mb][nb], false, false);
}

// ---------------------------------------------------------------------------
// Tiled bf16 WMMA GEMM:  C[M,N] = epi(A[M,K] @ W[K,N] + bias [+ res])
// f32 in memory, bf16 in LDS, f32 WMMA accumulate.
// 256 thr = 8 waves (2x4); WG tile 64x128; K-step 32.  M % 64 == 0.
// Double-buffered LDS + register prefetch: global b128 loads for tile t+1
// are in flight while tile t runs through the WMMA pipes; one barrier/step.
// ---------------------------------------------------------------------------
template<bool RELU, bool RES>
__global__ __launch_bounds__(256)
void gemm_wmma_bf16(const float* __restrict__ A, const float* __restrict__ W,
                    const float* __restrict__ bias, const float* __restrict__ res,
                    float* __restrict__ C, int M, int N, int K)
{
  __shared__ unsigned short As[2][64][36];   // pad->36: even, conflict-free
  __shared__ unsigned short Bs[2][128][36];  // transposed: Bs[buf][n][k]

  const int tid  = threadIdx.x;
  const int m0   = blockIdx.y * 64;
  const int n0   = blockIdx.x * 128;

  const int lane = tid & 31;
  const int wv   = tid >> 5;
  const int wm   = (wv >> 2) * 32;
  const int wn   = (wv & 3) * 32;
  const int lr   = lane & 15;
  const bool hi  = lane >= 16;

  // staging indices
  const int arow = tid >> 2;          // 0..63
  const int acol = (tid & 3) * 8;     // 0,8,16,24
  const int bn   = (tid & 31) * 4;    // 0..124  (4 consecutive n)
  const int bk   = (tid >> 5) * 2;    // 0..14   (pair of consecutive k)

  const bool nFull = (n0 + 128 <= N);
  const int  kFull = K & ~31;
  const int  nT    = kFull >> 5;

  const float* aBase = A + (size_t)(m0 + arow) * K + acol;

  float4   ra0, ra1;
  float4a4 rb0[2], rb1[2];

  auto fetch = [&](int k0) {
    const float* ap = aBase + k0;
    __builtin_prefetch(ap + 32, 0, 1);            // speculative global_prefetch
    ra0 = *(const float4*)ap;
    ra1 = *(const float4*)(ap + 4);
    if (nFull) {
#pragma unroll
      for (int p = 0; p < 2; ++p) {
        const float* wp = W + (size_t)(k0 + bk + p * 16) * N + n0 + bn;
        rb0[p] = *(const float4a4*)wp;
        rb1[p] = *(const float4a4*)(wp + N);
      }
    } else {
#pragma unroll
      for (int p = 0; p < 2; ++p) {
        const float* wp = W + (size_t)(k0 + bk + p * 16) * N;
#pragma unroll
        for (int j = 0; j < 4; ++j) {
          const int gn = n0 + bn + j;
          rb0[p][j] = (gn < N) ? wp[gn]     : 0.0f;
          rb1[p][j] = (gn < N) ? wp[N + gn] : 0.0f;
        }
      }
    }
  };

  auto stage = [&](int buf) {
    *(unsigned int*)&As[buf][arow][acol + 0] = pack_bf16x2(ra0.x, ra0.y);
    *(unsigned int*)&As[buf][arow][acol + 2] = pack_bf16x2(ra0.z, ra0.w);
    *(unsigned int*)&As[buf][arow][acol + 4] = pack_bf16x2(ra1.x, ra1.y);
    *(unsigned int*)&As[buf][arow][acol + 6] = pack_bf16x2(ra1.z, ra1.w);
#pragma unroll
    for (int p = 0; p < 2; ++p)
#pragma unroll
      for (int j = 0; j < 4; ++j)
        *(unsigned int*)&Bs[buf][bn + j][bk + p * 16] = pack_bf16x2(rb0[p][j], rb1[p][j]);
  };

  floatx8 acc[2][2] = {};

  if (nT > 0) fetch(0);
  int buf = 0;
  for (int t = 0; t < nT; ++t) {
    stage(buf);
    if (t + 1 < nT) fetch((t + 1) << 5);          // overlap with WMMA below
    __syncthreads();
    wmma_step(As[buf], Bs[buf], wm, wn, lr, hi, acc);
    buf ^= 1;
  }

  // ---- K tail (K % 32 != 0, e.g. EMB=300): guarded staging, one tile ----
  // 'buf' was last computed two steps ago; every wave passed the final
  // barrier after its compute on it, so staging here is race-free.
  if (kFull < K) {
    const float* arp = A + (size_t)(m0 + arow) * K;
#pragma unroll
    for (int j = 0; j < 8; j += 2) {
      const int kk = kFull + acol + j;
      const float v0 = (kk     < K) ? arp[kk]     : 0.0f;
      const float v1 = (kk + 1 < K) ? arp[kk + 1] : 0.0f;
      *(unsigned int*)&As[buf][arow][acol + j] = pack_bf16x2(v0, v1);
    }
#pragma unroll
    for (int p = 0; p < 2; ++p) {
      const int kk = kFull + bk + p * 16;
#pragma unroll
      for (int j = 0; j < 4; ++j) {
        const int gn = n0 + bn + j;
        const bool ok = (gn < N);
        const float v0 = (ok && kk     < K) ? W[(size_t)kk * N + gn]       : 0.0f;
        const float v1 = (ok && kk + 1 < K) ? W[(size_t)(kk + 1) * N + gn] : 0.0f;
        *(unsigned int*)&Bs[buf][bn + j][bk + p * 16] = pack_bf16x2(v0, v1);
      }
    }
    __syncthreads();
    wmma_step(As[buf], Bs[buf], wm, wn, lr, hi, acc);
  }

  // ---- epilogue: bias (+residual) (+relu) ----
#pragma unroll
  for (int mb = 0; mb < 2; ++mb) {
#pragma unroll
    for (int nb = 0; nb < 2; ++nb) {
      const int n = n0 + wn + nb * 16 + lr;
      if (n >= N) continue;
      const float bv = bias[n];
#pragma unroll
      for (int r = 0; r < 8; ++r) {
        const int m = m0 + wm + mb * 16 + r + (hi ? 8 : 0);
        float v = acc[mb][nb][r] + bv;
        if (RES) v += res[(size_t)m * N + n];
        if (RELU) v = fmaxf(v, 0.0f);
        C[(size_t)m * N + n] = v;
      }
    }
  }
}

// ---------------------------------------------------------------------------
// Auxiliary f32 kernels
// ---------------------------------------------------------------------------
__global__ void gather_kernel(const float* __restrict__ emb, const int* __restrict__ idx,
                              float* __restrict__ out, int dim)
{
  const int r = blockIdx.x;
  const int i = idx[r];
  for (int d = threadIdx.x; d < dim; d += blockDim.x)
    out[(size_t)r * dim + d] = emb[(size_t)i * dim + d];
}

__global__ void addpos_kernel(float* __restrict__ x, const float* __restrict__ pos, int L)
{
  const int b = blockIdx.y, l = blockIdx.x;
  float* xp = x + ((size_t)b * L + l) * HD;
  const float* pp = pos + (size_t)l * HD;
  for (int h = threadIdx.x; h < HD; h += blockDim.x) xp[h] += pp[h];
}

__global__ __launch_bounds__(256)
void layernorm_kernel(const float* __restrict__ x, const float* __restrict__ g,
                      const float* __restrict__ beta, float* __restrict__ y)
{
  const int row = blockIdx.x, tid = threadIdx.x;
  const float4 v = ((const float4*)(x + (size_t)row * HD))[tid];   // 256*4 = 1024
  __shared__ float red[256];
  red[tid] = v.x + v.y + v.z + v.w;
  __syncthreads();
  for (int s = 128; s > 0; s >>= 1) { if (tid < s) red[tid] += red[tid + s]; __syncthreads(); }
  const float mean = red[0] * (1.0f / HD);
  __syncthreads();
  const float dx = v.x - mean, dy = v.y - mean, dz = v.z - mean, dw = v.w - mean;
  red[tid] = dx * dx + dy * dy + dz * dz + dw * dw;
  __syncthreads();
  for (int s = 128; s > 0; s >>= 1) { if (tid < s) red[tid] += red[tid + s]; __syncthreads(); }
  const float r = rsqrtf(red[0] * (1.0f / HD) + 1e-6f);
  const float4 gv = ((const float4*)g)[tid];
  const float4 bv = ((const float4*)beta)[tid];
  float4 o;
  o.x = gv.x * dx * r + bv.x;  o.y = gv.y * dy * r + bv.y;
  o.z = gv.z * dz * r + bv.z;  o.w = gv.w * dw * r + bv.w;
  ((float4*)(y + (size_t)row * HD))[tid] = o;
}

// scores + mask + graph + softmax.  Q,K: [B, L, NH, DH].  att: [B, NH, Lq, Lk]
__global__ __launch_bounds__(256)
void attn_softmax_kernel(const float* __restrict__ Q, const float* __restrict__ K,
                         const float* __restrict__ mask, const float* __restrict__ graph,
                         float* __restrict__ att, int Lq, int Lk, float scale)
{
  const int b = blockIdx.z, h = blockIdx.y, qi = blockIdx.x, tid = threadIdx.x;
  __shared__ float qv[DHEAD];
  __shared__ float sc[272];
  __shared__ float red[256];
  const float* qp = Q + (((size_t)b * Lq + qi) * NHEAD + h) * DHEAD;
  if (tid < DHEAD) qv[tid] = qp[tid];
  __syncthreads();

  float lmax = -3.0e38f;
  for (int k = tid; k < Lk; k += 256) {
    const float* kp = K + (((size_t)b * Lk + k) * NHEAD + h) * DHEAD;
    float d = 0.0f;
#pragma unroll
    for (int j = 0; j < DHEAD; ++j) d += qv[j] * kp[j];
    float s = d * scale;
    if (mask[(size_t)b * Lk + k] <= 0.0f) s = -1e9f;
    if (graph && graph[((size_t)b * Lq + qi) * Lk + k] <= 0.0f) s = -1e9f;
    sc[k] = s;
    lmax = fmaxf(lmax, s);
  }
  red[tid] = lmax; __syncthreads();
  for (int s = 128; s > 0; s >>= 1) { if (tid < s) red[tid] = fmaxf(red[tid], red[tid + s]); __syncthreads(); }
  const float mx = red[0];
  __syncthreads();

  float lsum = 0.0f;
  for (int k = tid; k < Lk; k += 256) { const float e = __expf(sc[k] - mx); sc[k] = e; lsum += e; }
  red[tid] = lsum; __syncthreads();
  for (int s = 128; s > 0; s >>= 1) { if (tid < s) red[tid] += red[tid + s]; __syncthreads(); }
  const float inv = 1.0f / red[0];
  float* ap = att + (((size_t)b * NHEAD + h) * Lq + qi) * Lk;
  for (int k = tid; k < Lk; k += 256) ap[k] = sc[k] * inv;
}

// out[b,q,h,d] = sum_k att[b,h,q,k] * V[b,k,h,d]
__global__ __launch_bounds__(64)
void attn_apply_kernel(const float* __restrict__ att, const float* __restrict__ V,
                       float* __restrict__ out, int Lq, int Lk)
{
  const int b = blockIdx.z, h = blockIdx.y, qi = blockIdx.x, d = threadIdx.x;
  const float* ap = att + (((size_t)b * NHEAD + h) * Lq + qi) * Lk;
  float acc = 0.0f;
  for (int k = 0; k < Lk; ++k)
    acc += ap[k] * V[(((size_t)b * Lk + k) * NHEAD + h) * DHEAD + d];
  out[(((size_t)b * Lq + qi) * NHEAD + h) * DHEAD + d] = acc;
}

__global__ void catcopy_kernel(const float* __restrict__ src, float* __restrict__ dst,
                               int Ls, int Ld, int off)
{
  const int b = blockIdx.y, l = blockIdx.x;
  const float* s = src + ((size_t)b * Ls + l) * HD;
  float* d = dst + ((size_t)b * Ld + off + l) * HD;
  for (int h = threadIdx.x; h < HD; h += blockDim.x) d[h] = s[h];
}

__global__ void catmask_kernel(const float* __restrict__ m1, const float* __restrict__ m2,
                               float* __restrict__ out, int L1, int L2)
{
  const int b = blockIdx.x;
  for (int i = threadIdx.x; i < L1 + L2; i += blockDim.x)
    out[(size_t)b * (L1 + L2) + i] = (i < L1) ? m1[(size_t)b * L1 + i]
                                              : m2[(size_t)b * L2 + (i - L1)];
}

__global__ void mulmask_kernel(float* __restrict__ x, const float* __restrict__ mask)
{
  const int r = blockIdx.x;
  const float m = mask[r];
  float* xp = x + (size_t)r * HD;
  for (int h = threadIdx.x; h < HD; h += blockDim.x) xp[h] *= m;
}

__global__ void maxpool_kernel(const float* __restrict__ x, const float* __restrict__ mask,
                               float* __restrict__ out, int L)
{
  const int b = blockIdx.x;
  for (int h = threadIdx.x; h < HD; h += blockDim.x) {
    float m = -3.0e38f;
    for (int l = 0; l < L; ++l)
      m = fmaxf(m, x[((size_t)b * L + l) * HD + h] * mask[(size_t)b * L + l]);
    out[(size_t)b * HD + h] = m;
  }
}

// ---------------------------------------------------------------------------
// Host-side parameter walking (jax tree_flatten: dict keys sorted; lists kept)
// ---------------------------------------------------------------------------
struct Lin { const float* b; const float* w; };
struct LNp { const float* beta; const float* g; };
struct Att { Lin k; LNp ln; Lin o; Lin q; Lin v; };
struct Ffn { LNp ln; Lin w1; Lin w2; };
struct Stack { Att att[NBLK]; Ffn ffn[NBLK]; };

struct Walker {
  const float* p; size_t off = 0;
  const float* take(size_t n) { const float* r = p + off; off += n; return r; }
  Lin lin(int din, int dout) { Lin l; l.b = take(dout); l.w = take((size_t)din * dout); return l; }
  LNp ln() { LNp l; l.beta = take(HD); l.g = take(HD); return l; }
  Att att() { Att a; a.k = lin(HD, HD); a.ln = ln(); a.o = lin(HD, HD); a.q = lin(HD, HD); a.v = lin(HD, HD); return a; }
  Ffn ffn() { Ffn f; f.ln = ln(); f.w1 = lin(HD, FFH); f.w2 = lin(FFH, HD); return f; }
  Stack stack() { Stack s; for (int i = 0; i < NBLK; ++i) s.att[i] = att();
                  for (int i = 0; i < NBLK; ++i) s.ffn[i] = ffn(); return s; }
};

static inline void gemm(hipStream_t st, const float* A, const Lin& L, const float* res,
                        float* C, int M, int N, int K, bool relu, bool resid)
{
  dim3 g((N + 127) / 128, M / 64), b(256);
  if (relu)       gemm_wmma_bf16<true , false><<<g, b, 0, st>>>(A, L.w, L.b, nullptr, C, M, N, K);
  else if (resid) gemm_wmma_bf16<false, true ><<<g, b, 0, st>>>(A, L.w, L.b, res,     C, M, N, K);
  else            gemm_wmma_bf16<false, false><<<g, b, 0, st>>>(A, L.w, L.b, nullptr, C, M, N, K);
}

// self-attention encoder stack (4 blocks)
static void run_self_stack(hipStream_t st, const Stack& S, float* x, int L,
                           const float* mask, const float* graph,
                           float* att_out,       // nullptr -> scratch
                           float* Qb, float* Kb, float* Vb, float* ctx, float* big)
{
  const int M = BB * L;
  for (int blk = 0; blk < NBLK; ++blk) {
    const Att& A = S.att[blk];
    gemm(st, x, A.q, nullptr, Qb, M, HD, HD, false, false);
    gemm(st, x, A.k, nullptr, Kb, M, HD, HD, false, false);
    gemm(st, x, A.v, nullptr, Vb, M, HD, HD, false, false);
    float* att = att_out ? (att_out + (size_t)blk * BB * NHEAD * L * L) : big;
    attn_softmax_kernel<<<dim3(L, NHEAD, BB), 256, 0, st>>>(Qb, Kb, mask, graph, att, L, L, 0.125f);
    attn_apply_kernel<<<dim3(L, NHEAD, BB), 64, 0, st>>>(att, Vb, ctx, L, L);
    gemm(st, ctx, A.o, x, Qb, M, HD, HD, false, true);          // o-proj + residual
    layernorm_kernel<<<M, 256, 0, st>>>(Qb, A.ln.g, A.ln.beta, x);
    const Ffn& F = S.ffn[blk];
    gemm(st, x, F.w1, nullptr, big, M, FFH, HD, true, false);   // relu
    gemm(st, big, F.w2, x, Qb, M, HD, FFH, false, true);        // + residual
    layernorm_kernel<<<M, 256, 0, st>>>(Qb, F.ln.g, F.ln.beta, x);
  }
}

// ---------------------------------------------------------------------------
// kernel_launch: full forward pass
// ---------------------------------------------------------------------------
extern "C" void kernel_launch(void* const* d_in, const int* in_sizes, int n_in,
                              void* d_out, int out_size, void* d_ws, size_t ws_size,
                              hipStream_t stream)
{
  const float* vis_fea  = (const float*)d_in[0];   // [B,LV,2048]
  const float* vis_mask = (const float*)d_in[1];   // [B,LV]
  const int*   q_ipt    = (const int*)  d_in[2];   // [B,LQ]
  const float* q_mask   = (const float*)d_in[3];   // [B,LQ]
  /* d_in[4] = q_graph, unused (matches reference) */
  const int*   syb_ipt  = (const int*)  d_in[5];   // [B,LS]
  const float* syb_mask = (const float*)d_in[6];   // [B,LS]
  const float* syb_grph = (const float*)d_in[7];   // [B,LS,LS]

  // ---- params: top-level keys sorted alphabetically ----
  Walker W{ (const float*)d_in[8] };
  Lin cls0 = W.lin(HD, HD);
  Lin cls1 = W.lin(HD, NCLSS);
  Stack dec_s = W.stack();
  Stack mil_s = W.stack();
  const float* q_emb = W.take((size_t)50000 * EMBB);
  Lin q_mlp0 = W.lin(EMBB, HD);
  Lin q_mlp1 = W.lin(HD, HD);
  const float* q_pos = W.take((size_t)LQQ * HD);
  Stack syb_s = W.stack();
  const float* syb_emb = W.take((size_t)50000 * EMBB);
  Lin syb_mlp0 = W.lin(EMBB, 2048);
  Lin syb_mlp1 = W.lin(2048, HD);
  const float* syb_pos = W.take((size_t)LSS * HD);
  Lin v_mlp0 = W.lin(2048, HD);
  Lin v_mlp1 = W.lin(HD, HD);
  const float* v_pos = W.take((size_t)LVV * HD);
  Stack vis_s = W.stack();

  // ---- workspace layout (floats); total ~173M floats (~692 MB) ----
  float* ws = (float*)d_ws;
  size_t o = 0;
  auto alloc = [&](size_t n) { float* p = ws + o; o += n; return p; };
  float* visx = alloc((size_t)BB * LVV * HD);
  float* sybx = alloc((size_t)BB * LSS * HD);
  float* milx = alloc((size_t)BB * LTT * HD);
  float* decx = alloc((size_t)BB * LQQ * HD);
  float* Qb   = alloc((size_t)BB * LTT * HD);
  float* Kb   = alloc((size_t)BB * LTT * HD);
  float* Vb   = alloc((size_t)BB * LTT * HD);
  float* ctx  = alloc((size_t)BB * LTT * HD);
  float* big  = alloc((size_t)BB * LTT * FFH);   // FFN hidden; aliases attn-score scratch
  float* cmsk = alloc((size_t)BB * LTT);
  float* pool = alloc((size_t)BB * HD);
  float* tcls = alloc((size_t)BB * HD);
  float* gbuf = alloc((size_t)BB * LSS * EMBB);

  float* logits  = (float*)d_out;
  float* att_out = logits + (size_t)BB * NCLSS;   // 4 x [B,NH,260,260]

  // ---- question branch: dec = mlp2(q_mlp, q_emb[q_ipt]) + q_pos ----
  gather_kernel<<<BB * LQQ, 256, 0, stream>>>(q_emb, q_ipt, gbuf, EMBB);
  gemm(stream, gbuf, q_mlp0, nullptr, Qb, BB * LQQ, HD, EMBB, true, false);
  gemm(stream, Qb, q_mlp1, nullptr, decx, BB * LQQ, HD, HD, false, false);
  addpos_kernel<<<dim3(LQQ, BB), 256, 0, stream>>>(decx, q_pos, LQQ);

  // ---- vision branch: fea_vis = mlp2(v_mlp, vis_fea) + v_pos; encode ----
  gemm(stream, vis_fea, v_mlp0, nullptr, Qb, BB * LVV, HD, 2048, true, false);
  gemm(stream, Qb, v_mlp1, nullptr, visx, BB * LVV, HD, HD, false, false);
  addpos_kernel<<<dim3(LVV, BB), 256, 0, stream>>>(visx, v_pos, LVV);
  run_self_stack(stream, vis_s, visx, LVV, vis_mask, nullptr, nullptr, Qb, Kb, Vb, ctx, big);

  // ---- symbol branch: syb = mlp2(syb_mlp, syb_emb[syb_ipt]) + syb_pos; graph encode ----
  gather_kernel<<<BB * LSS, 256, 0, stream>>>(syb_emb, syb_ipt, gbuf, EMBB);
  gemm(stream, gbuf, syb_mlp0, nullptr, Kb, BB * LSS, 2048, EMBB, true, false);
  gemm(stream, Kb, syb_mlp1, nullptr, sybx, BB * LSS, HD, 2048, false, false);
  addpos_kernel<<<dim3(LSS, BB), 256, 0, stream>>>(sybx, syb_pos, LSS);
  run_self_stack(stream, syb_s, sybx, LSS, syb_mask, syb_grph, nullptr, Qb, Kb, Vb, ctx, big);

  // ---- MIL: concat, encode with attention collection into d_out ----
  catmask_kernel<<<BB, 256, 0, stream>>>(vis_mask, syb_mask, cmsk, LVV, LSS);
  catcopy_kernel<<<dim3(LVV, BB), 256, 0, stream>>>(visx, milx, LVV, LTT, 0);
  catcopy_kernel<<<dim3(LSS, BB), 256, 0, stream>>>(sybx, milx, LSS, LTT, LVV);
  run_self_stack(stream, mil_s, milx, LTT, cmsk, nullptr, att_out, Qb, Kb, Vb, ctx, big);
  mulmask_kernel<<<BB * LTT, 256, 0, stream>>>(milx, cmsk);

  // ---- decoder: cross-attention blocks ----
  for (int blk = 0; blk < NBLK; ++blk) {
    const Att& A = dec_s.att[blk];
    gemm(stream, decx, A.q, nullptr, Qb, BB * LQQ, HD, HD, false, false);
    gemm(stream, milx, A.k, nullptr, Kb, BB * LTT, HD, HD, false, false);
    gemm(stream, milx, A.v, nullptr, Vb, BB * LTT, HD, HD, false, false);
    attn_softmax_kernel<<<dim3(LQQ, NHEAD, BB), 256, 0, stream>>>(Qb, Kb, cmsk, nullptr, big, LQQ, LTT, 0.125f);
    attn_apply_kernel<<<dim3(LQQ, NHEAD, BB), 64, 0, stream>>>(big, Vb, ctx, LQQ, LTT);
    gemm(stream, ctx, A.o, decx, Qb, BB * LQQ, HD, HD, false, true);
    layernorm_kernel<<<BB * LQQ, 256, 0, stream>>>(Qb, A.ln.g, A.ln.beta, decx);
    const Ffn& F = dec_s.ffn[blk];
    gemm(stream, decx, F.w1, nullptr, big, BB * LQQ, FFH, HD, true, false);
    gemm(stream, big, F.w2, decx, Qb, BB * LQQ, HD, FFH, false, true);
    layernorm_kernel<<<BB * LQQ, 256, 0, stream>>>(Qb, F.ln.g, F.ln.beta, decx);
  }

  // ---- mask, max-pool over Lq, classifier ----
  maxpool_kernel<<<BB, 256, 0, stream>>>(decx, q_mask, pool, LQQ);
  gemm(stream, pool, cls0, nullptr, tcls, BB, HD, HD, true, false);
  gemm(stream, tcls, cls1, nullptr, logits, BB, NCLSS, HD, false, false);
}
